// MultiHeadSelfAttentionNaive_72267119722913
// MI455X (gfx1250) — matrix operand order repaired and checked by reading, not compiled, not run on hardware
//
#include <hip/hip_runtime.h>

typedef __attribute__((ext_vector_type(16))) _Float16 v16h;
typedef __attribute__((ext_vector_type(8)))  _Float16 v8h;
typedef __attribute__((ext_vector_type(8)))  float    v8f;
typedef __attribute__((ext_vector_type(4)))  int      v4i;

#define WMMA_F32_F16(a, b, c) \
  __builtin_amdgcn_wmma_f32_16x16x32_f16(false, (a), false, (b), (short)0, (c), false, false)

// ---- CDNA5 async global->LDS copy (ASYNCcnt path), guarded ----------------
#ifndef __has_builtin
#define __has_builtin(x) 0
#endif
#if __has_builtin(__builtin_amdgcn_global_load_async_to_lds_b128)
#define HAVE_ASYNC_COPY 1
#else
#define HAVE_ASYNC_COPY 0
#endif

typedef __attribute__((address_space(1))) v4i gv4i;  // global int4
typedef __attribute__((address_space(3))) v4i lv4i;  // LDS int4

static __device__ __forceinline__ void copy_b128(const _Float16* g, _Float16* l) {
#if HAVE_ASYNC_COPY
  __builtin_amdgcn_global_load_async_to_lds_b128((gv4i*)g, (lv4i*)l, 0, 0);
#else
  *(v8h*)l = *(const v8h*)g;
#endif
}

static __device__ __forceinline__ void async_wait0() {
#if HAVE_ASYNC_COPY
#if __has_builtin(__builtin_amdgcn_s_wait_asynccnt)
  __builtin_amdgcn_s_wait_asynccnt(0);
#else
  asm volatile("s_wait_asynccnt 0x0" ::: "memory");
#endif
#endif
}

static __device__ __forceinline__ v16h cat16(v8h lo, v8h hi) {
  return __builtin_shufflevector(lo, hi, 0,1,2,3,4,5,6,7,8,9,10,11,12,13,14,15);
}

// ---------------------------------------------------------------------------
// GEMM: C[M x Nfull] = A[M x K] * W[K x Nfull] + bias
// Block tile 128x128, BK=32, 256 threads (8 waves), wave tile 64x32 (4x2 WMMA).
// mode 0: scatter f16 into q/k/v (B,H,N,D).   mode 1: f32 row-major output.
// AHALF: A is f16 (attention output, async-copied); else A is f32 (converted).
// ---------------------------------------------------------------------------
template <bool AHALF>
__global__ __launch_bounds__(256) void gemm_wmma(
    const void* __restrict__ Aptr, const float* __restrict__ W,
    const float* __restrict__ bias,
    _Float16* __restrict__ qd, _Float16* __restrict__ kd, _Float16* __restrict__ vd,
    float* __restrict__ outf, int Nfull, int Kdim, int mode) {
  constexpr int LDA = 40;  // halves; byte stride 80 (16B aligned)
  __shared__ _Float16 As[128 * LDA];
  __shared__ _Float16 Bs[128 * LDA];  // Bs transposed: [col][k]

  const int tid   = threadIdx.x;
  const int lane  = tid & 31;
  const int w     = tid >> 5;
  const int waveM = w >> 2;       // 0..1
  const int waveN = w & 3;        // 0..3
  const int lrow  = lane & 15;
  const int hi    = lane >> 4;    // 0/1
  const int kbA   = hi * 8;       // A fragment K base
  const int kbB   = hi * 8;       // B fragment K base

  const int rowBase = blockIdx.y * 128;
  const int colBase = blockIdx.x * 128;

  v8f acc[4][2];
  const v8f vzero = {0.f, 0.f, 0.f, 0.f, 0.f, 0.f, 0.f, 0.f};
#pragma unroll
  for (int tm = 0; tm < 4; ++tm)
#pragma unroll
    for (int tn = 0; tn < 2; ++tn) acc[tm][tn] = vzero;

  for (int k0 = 0; k0 < Kdim; k0 += 32) {
    __syncthreads();
    if (AHALF) {
      const _Float16* Ah = (const _Float16*)Aptr;
      const int kg = tid & 3, r0 = tid >> 2;  // 8 halves per copy, 64 rows
#pragma unroll
      for (int i = 0; i < 2; ++i) {
        const int row = r0 + i * 64;
        copy_b128(Ah + (size_t)(rowBase + row) * Kdim + k0 + kg * 8,
                  &As[row * LDA + kg * 8]);
      }
    } else {
      const float* Af = (const float*)Aptr;
      const int kg = tid & 7, r0 = tid >> 3;  // float4 per load, 32 rows
#pragma unroll
      for (int i = 0; i < 4; ++i) {
        const int row = r0 + i * 32;
        const float4 f = *(const float4*)(Af + (size_t)(rowBase + row) * Kdim + k0 + kg * 4);
        _Float16* dst = &As[row * LDA + kg * 4];
        dst[0] = (_Float16)f.x; dst[1] = (_Float16)f.y;
        dst[2] = (_Float16)f.z; dst[3] = (_Float16)f.w;
      }
    }
    {  // W tile: convert f32->f16, store transposed [col][k]
      const int kk = tid >> 5, ng = tid & 31;
#pragma unroll
      for (int i = 0; i < 4; ++i) {
        const int krow = kk + i * 8;
        const float4 f = *(const float4*)(W + (size_t)(k0 + krow) * Nfull + colBase + ng * 4);
        Bs[(ng * 4 + 0) * LDA + krow] = (_Float16)f.x;
        Bs[(ng * 4 + 1) * LDA + krow] = (_Float16)f.y;
        Bs[(ng * 4 + 2) * LDA + krow] = (_Float16)f.z;
        Bs[(ng * 4 + 3) * LDA + krow] = (_Float16)f.w;
      }
    }
    if (AHALF) async_wait0();
    __syncthreads();

    v16h afrag[4];
#pragma unroll
    for (int tm = 0; tm < 4; ++tm) {
      const _Float16* p = &As[(waveM * 64 + tm * 16 + lrow) * LDA + kbA];
      afrag[tm] = cat16(*(const v8h*)p, *(const v8h*)(p + 16));
    }
    v16h bfrag[2];
#pragma unroll
    for (int tn = 0; tn < 2; ++tn) {
      const _Float16* p = &Bs[(waveN * 32 + tn * 16 + lrow) * LDA + kbB];
      bfrag[tn] = cat16(*(const v8h*)p, *(const v8h*)(p + 16));
    }
#pragma unroll
    for (int tm = 0; tm < 4; ++tm)
#pragma unroll
      for (int tn = 0; tn < 2; ++tn)
        acc[tm][tn] = WMMA_F32_F16(afrag[tm], bfrag[tn], acc[tm][tn]);
  }

  // Epilogue
#pragma unroll
  for (int tm = 0; tm < 4; ++tm) {
#pragma unroll
    for (int tn = 0; tn < 2; ++tn) {
      const int gcol = colBase + waveN * 32 + tn * 16 + lrow;
      const float bv = bias[gcol];
#pragma unroll
      for (int r = 0; r < 8; ++r) {
        const int grow = rowBase + waveM * 64 + tm * 16 + hi * 8 + r;
        const float val = acc[tm][tn][r] + bv;
        if (mode == 0) {
          const int which = gcol >> 10;       // 0:q 1:k 2:v
          const int c1 = gcol & 1023;
          const int hh = c1 >> 6, dd = c1 & 63;
          const int bb = grow >> 11, nn = grow & 2047;
          _Float16* dst = (which == 0) ? qd : (which == 1) ? kd : vd;
          dst[((size_t)(bb * 16 + hh) * 2048 + nn) * 64 + dd] = (_Float16)val;
        } else {
          outf[(size_t)grow * Nfull + gcol] = val;
        }
      }
    }
  }
}

// ---------------------------------------------------------------------------
// Flash attention: one block (4 waves, 128 thr) per (b*h, 64 query rows).
// Q/K tiles async-copied to LDS (ASYNCcnt), V transposed via VALU,
// S = Q*K^T (WMMA), online softmax (half-wave shuffles), P staged in
// wave-private LDS (s_wait_dscnt fence), O += P*V (WMMA). Output f16 (B,N,C).
// ---------------------------------------------------------------------------
__global__ __launch_bounds__(128) void attn_wmma(
    const _Float16* __restrict__ q, const _Float16* __restrict__ kx,
    const _Float16* __restrict__ vx, _Float16* __restrict__ o) {
  constexpr int LD = 72;  // halves; byte stride 144 (16B aligned)
  __shared__ _Float16 Qs[64 * LD];
  __shared__ _Float16 Ks[64 * LD];       // [key][d]
  __shared__ _Float16 VsT[64 * LD];      // [d][key]
  __shared__ _Float16 Ps[4 * 16 * LD];   // per-wave 16x64 P tile

  const int tid  = threadIdx.x;
  const int lane = tid & 31;
  const int w    = tid >> 5;   // 0..3
  const int lrow = lane & 15;
  const int hi   = lane >> 4;
  const int kbA  = hi * 8;
  const int kbB  = hi * 8;

  const int bh   = blockIdx.y;
  const int qblk = blockIdx.x;
  const size_t base = (size_t)bh * 2048 * 64;
  const _Float16* qb = q + base + (size_t)qblk * 64 * 64;

  {  // load Q tile (64x64) via async copy
    const int dg = tid & 7, r0 = tid >> 3;
#pragma unroll
    for (int i = 0; i < 4; ++i) {
      const int row = r0 + i * 16;
      copy_b128(qb + (size_t)row * 64 + dg * 8, &Qs[row * LD + dg * 8]);
    }
  }
  async_wait0();
  __syncthreads();

  v16h aq[2];  // Q fragments for d = 0..31, 32..63 (constant across key blocks)
#pragma unroll
  for (int kk = 0; kk < 2; ++kk) {
    const _Float16* p = &Qs[(w * 16 + lrow) * LD + kk * 32 + kbA];
    aq[kk] = cat16(*(const v8h*)p, *(const v8h*)(p + 16));
  }

  const v8f vzero = {0.f, 0.f, 0.f, 0.f, 0.f, 0.f, 0.f, 0.f};
  float m[8], l[8];
  v8f oacc[4];
#pragma unroll
  for (int r = 0; r < 8; ++r) { m[r] = -3.0e30f; l[r] = 0.f; }
#pragma unroll
  for (int td = 0; td < 4; ++td) oacc[td] = vzero;

  for (int j = 0; j < 32; ++j) {
    __syncthreads();  // protect Ks/VsT reuse
    const _Float16* kt = kx + base + (size_t)j * 64 * 64;
    const _Float16* vt = vx + base + (size_t)j * 64 * 64;
    const int dg = tid & 7, r0 = tid >> 3;
#pragma unroll
    for (int i = 0; i < 4; ++i) {
      const int row = r0 + i * 16;
      copy_b128(kt + (size_t)row * 64 + dg * 8, &Ks[row * LD + dg * 8]);  // async
      v8h tv = *(const v8h*)(vt + (size_t)row * 64 + dg * 8);
#pragma unroll
      for (int e = 0; e < 8; ++e) VsT[(dg * 8 + e) * LD + row] = tv[e];
    }
    async_wait0();
    __syncthreads();

    // S = Q * K^T  (16 query rows x 64 keys per wave)
    v8f s[4];
#pragma unroll
    for (int tn = 0; tn < 4; ++tn) s[tn] = vzero;
#pragma unroll
    for (int kk = 0; kk < 2; ++kk) {
#pragma unroll
      for (int tn = 0; tn < 4; ++tn) {
        const _Float16* p = &Ks[(tn * 16 + lrow) * LD + kk * 32 + kbB];
        v16h bk = cat16(*(const v8h*)p, *(const v8h*)(p + 16));
        s[tn] = WMMA_F32_F16(aq[kk], bk, s[tn]);
      }
    }

    // online softmax per row (row values live on 16 lanes of a half-wave)
#pragma unroll
    for (int r = 0; r < 8; ++r) {
      float mc = fmaxf(fmaxf(s[0][r], s[1][r]), fmaxf(s[2][r], s[3][r]));
#pragma unroll
      for (int off = 1; off < 16; off <<= 1) mc = fmaxf(mc, __shfl_xor(mc, off));
      const float mn = fmaxf(m[r], mc);
      const float sc = __expf(m[r] - mn);
      m[r] = mn;
      float rs = 0.f;
#pragma unroll
      for (int tn = 0; tn < 4; ++tn) {
        const float p0 = __expf(s[tn][r] - mn);
        s[tn][r] = p0;
        rs += p0;
      }
#pragma unroll
      for (int off = 1; off < 16; off <<= 1) rs += __shfl_xor(rs, off);
      l[r] = l[r] * sc + rs;
#pragma unroll
      for (int td = 0; td < 4; ++td) oacc[td][r] *= sc;
    }

    // stage P (f16) into wave-private LDS, then reuse as A fragments
    _Float16* Pw = &Ps[w * 16 * LD];
#pragma unroll
    for (int tn = 0; tn < 4; ++tn)
#pragma unroll
      for (int r = 0; r < 8; ++r)
        Pw[(hi * 8 + r) * LD + tn * 16 + lrow] = (_Float16)s[tn][r];
    asm volatile("s_wait_dscnt 0" ::: "memory");  // wave-level LDS RAW fence

    // O += P * V
#pragma unroll
    for (int kk = 0; kk < 2; ++kk) {
      const _Float16* pp = &Pw[lrow * LD + kk * 32 + kbA];
      v16h ap = cat16(*(const v8h*)pp, *(const v8h*)(pp + 16));
#pragma unroll
      for (int td = 0; td < 4; ++td) {
        const _Float16* pv = &VsT[(td * 16 + lrow) * LD + kk * 32 + kbB];
        v16h bv = cat16(*(const v8h*)pv, *(const v8h*)(pv + 16));
        oacc[td] = WMMA_F32_F16(ap, bv, oacc[td]);
      }
    }
  }

  // normalize and store (B,N,C) f16, c = h*64 + d
  const int b = bh >> 4, h = bh & 15;
#pragma unroll
  for (int td = 0; td < 4; ++td) {
    const int d = td * 16 + lrow;
#pragma unroll
    for (int r = 0; r < 8; ++r) {
      const int n = qblk * 64 + w * 16 + hi * 8 + r;
      const float val = oacc[td][r] / l[r];
      o[((size_t)(b * 2048 + n)) * 1024 + h * 64 + d] = (_Float16)val;
    }
  }
}

// ---------------------------------------------------------------------------
extern "C" void kernel_launch(void* const* d_in, const int* in_sizes, int n_in,
                              void* d_out, int out_size, void* d_ws, size_t ws_size,
                              hipStream_t stream) {
  const float* x     = (const float*)d_in[0];
  const float* w_in  = (const float*)d_in[1];
  const float* b_in  = (const float*)d_in[2];
  const float* w_out = (const float*)d_in[3];
  const float* b_out = (const float*)d_in[4];
  float* out = (float*)d_out;

  const size_t QKV = (size_t)8192 * 1024;  // elements per q/k/v tensor
  _Float16* qh = (_Float16*)d_ws;
  _Float16* kh = qh + QKV;
  _Float16* vh = kh + QKV;
  _Float16* oh = vh + QKV;  // attention output, f16 (B,N,C)

  // 1) qkv = x @ w_in + b_in  -> scattered f16 q/k/v (B,H,N,D)
  gemm_wmma<false><<<dim3(3072 / 128, 8192 / 128), 256, 0, stream>>>(
      x, w_in, b_in, qh, kh, vh, nullptr, 3072, 1024, 0);

  // 2) flash attention per (b,h), 64 query rows per block
  attn_wmma<<<dim3(2048 / 64, 4 * 16), 128, 0, stream>>>(qh, kh, vh, oh);

  // 3) out = attn @ w_out + b_out -> f32
  gemm_wmma<true><<<dim3(1024 / 128, 8192 / 128), 256, 0, stream>>>(
      oh, w_out, b_out, nullptr, nullptr, nullptr, out, 1024, 1024, 1);
}
